// TemporalAttention_50861002719348
// MI455X (gfx1250) — matrix operand order repaired and checked by reading, not compiled, not run on hardware
//
#include <hip/hip_runtime.h>
#include <hip/hip_bf16.h>

// ---------------------------------------------------------------------------
// TemporalAttention for MI455X (gfx1250, wave32, WMMA)
// qkv proj (WMMA GEMM) -> flash attention (WMMA QK^T + PV) -> out proj (WMMA)
// f16 storage, f32 accumulation via v_wmma_f32_16x16x32_f16.
// GEMM staging via GLOBAL_LOAD_ASYNC_TO_LDS (double-buffered) when the
// toolchain exposes the builtin; else register-prefetch pipeline.
// ---------------------------------------------------------------------------

#define D_MODEL 1024
#define N_HEADS 16
#define D_HEAD  64
#define BATCH   4
#define SEQ     2048
#define QKV_N   (3 * D_MODEL)
#define TOKENS  (BATCH * SEQ)

typedef __attribute__((ext_vector_type(4)))  int      v4i;
typedef __attribute__((ext_vector_type(16))) _Float16 v16h;
typedef __attribute__((ext_vector_type(8)))  float    v8f;

#if defined(__has_builtin)
#if __has_builtin(__builtin_amdgcn_global_load_async_to_lds_b128)
#define TA_ASYNC 1
#endif
#endif
#ifndef TA_ASYNC
#define TA_ASYNC 0
#endif

#if TA_ASYNC
// signature (from hipcc diagnostic): (v4i AS1* src, v4i AS3* dst, imm, imm)
#define TA_ASYNC_COPY_B128(gsrc, ldst)                                        \
    __builtin_amdgcn_global_load_async_to_lds_b128(                           \
        (__attribute__((address_space(1))) v4i*)(void*)(gsrc),                \
        (__attribute__((address_space(3))) v4i*)(void*)(ldst), 0, 0)
#if __has_builtin(__builtin_amdgcn_s_wait_asynccnt)
#define TA_WAIT_ASYNC(n) __builtin_amdgcn_s_wait_asynccnt(n)
#else
#define TA_WAIT_ASYNC(n) asm volatile("s_wait_asynccnt " #n ::: "memory")
#endif
#endif

struct Frag {
    union {
        v16h     v;
        _Float16 h[16];
        uint4    q[2];
    };
};

union HalfPack8 {
    uint4    q;
    _Float16 h[8];
};

__device__ __forceinline__ float rowmax16(float v) {
    #pragma unroll
    for (int m = 1; m < 16; m <<= 1) v = fmaxf(v, __shfl_xor(v, m, 32));
    return v;
}
__device__ __forceinline__ float rowsum16(float v) {
    #pragma unroll
    for (int m = 1; m < 16; m <<= 1) v += __shfl_xor(v, m, 32);
    return v;
}

// ---------------------------------------------------------------------------
// f32 -> f16 cast, 4 elements / thread (float4 in, 8B packed out)
// ---------------------------------------------------------------------------
__global__ void ta_cast_f32_f16(const float* __restrict__ src,
                                _Float16* __restrict__ dst, int n4) {
    int i = blockIdx.x * 256 + threadIdx.x;
    if (i < n4) {
        float4 f = ((const float4*)src)[i];
        union { uint2 u; _Float16 h[4]; } p;
        p.h[0] = (_Float16)f.x; p.h[1] = (_Float16)f.y;
        p.h[2] = (_Float16)f.z; p.h[3] = (_Float16)f.w;
        ((uint2*)dst)[i] = p.u;
    }
}

// ---------------------------------------------------------------------------
// GEMM: Y[m,n] = sum_k A[m,k] * Bw[n,k] + bias[n]
// A: M x K (f16, row major), Bw: N x K (f16, row major).
// Block tile 128x128, 256 threads = 8 waves (4 in M x 2 in N),
// wave tile 32x64 (2x4 WMMA accumulators). K-step 32.
// If outH != null store f16, else f32 to outF. M,N % 128 == 0, K % 32 == 0.
// ---------------------------------------------------------------------------
__global__ __launch_bounds__(256)
void ta_gemm_f16(const _Float16* __restrict__ A,
                 const _Float16* __restrict__ Bw,
                 const float* __restrict__ bias,
                 float* __restrict__ outF,
                 _Float16* __restrict__ outH,
                 int M, int N, int K) {
    const int tid  = threadIdx.x;
    const int wave = tid >> 5;
    const int lane = tid & 31;
    const int ln   = lane & 15;
    const int hi   = lane >> 4;
    const int wm   = wave & 3;   // rows wm*32 .. +31
    const int wn   = wave >> 2;  // cols wn*64 .. +63

    const int n0 = blockIdx.x * 128;
    const int m0 = blockIdx.y * 128;

    // staging: 16 halves (2 x b128) per thread per matrix
    const int srow = tid >> 1;        // 0..127
    const int scol = (tid & 1) * 16;  // 0 or 16

    const _Float16* aptr = A  + (size_t)(m0 + srow) * K + scol;
    const _Float16* bptr = Bw + (size_t)(n0 + srow) * K + scol;

    v8f acc[2][4];
    #pragma unroll
    for (int i = 0; i < 2; ++i)
        #pragma unroll
        for (int c = 0; c < 4; ++c) acc[i][c] = (v8f)0.0f;

#if TA_ASYNC
    // ---------------- async double-buffered staging ----------------
    __shared__ __align__(16) _Float16 As[2][128][32];
    __shared__ __align__(16) _Float16 Bs[2][128][32];

    // stage 0 -> buffer 0 (4 async ops per wave)
    TA_ASYNC_COPY_B128(aptr,     &As[0][srow][scol]);
    TA_ASYNC_COPY_B128(aptr + 8, &As[0][srow][scol + 8]);
    TA_ASYNC_COPY_B128(bptr,     &Bs[0][srow][scol]);
    TA_ASYNC_COPY_B128(bptr + 8, &Bs[0][srow][scol + 8]);

    int buf = 0;
    for (int k0 = 0; k0 < K; k0 += 32, buf ^= 1) {
        if (k0 + 32 < K) {
            // background-fill the other buffer (freed by last stage's barrier)
            TA_ASYNC_COPY_B128(aptr + k0 + 32, &As[buf ^ 1][srow][scol]);
            TA_ASYNC_COPY_B128(aptr + k0 + 40, &As[buf ^ 1][srow][scol + 8]);
            TA_ASYNC_COPY_B128(bptr + k0 + 32, &Bs[buf ^ 1][srow][scol]);
            TA_ASYNC_COPY_B128(bptr + k0 + 40, &Bs[buf ^ 1][srow][scol + 8]);
            TA_WAIT_ASYNC(4);   // in-order: current buffer's 4 ops are done
        } else {
            TA_WAIT_ASYNC(0);
        }
        __syncthreads();        // all waves' copies for this buffer landed

        Frag af[2];
        #pragma unroll
        for (int i = 0; i < 2; ++i) {
            const int row = wm * 32 + i * 16 + ln;
            af[i].q[0] = *(const uint4*)&As[buf][row][8 * hi];
            af[i].q[1] = *(const uint4*)&As[buf][row][16 + 8 * hi];
        }
        Frag bf[4];
        #pragma unroll
        for (int c = 0; c < 4; ++c) {
            const int n = wn * 64 + c * 16 + ln;
            bf[c].q[0] = *(const uint4*)&Bs[buf][n][8 * hi];
            bf[c].q[1] = *(const uint4*)&Bs[buf][n][16 + 8 * hi];
        }
        #pragma unroll
        for (int i = 0; i < 2; ++i)
            #pragma unroll
            for (int c = 0; c < 4; ++c)
                acc[i][c] = __builtin_amdgcn_wmma_f32_16x16x32_f16(
                    false, af[i].v, false, bf[c].v, (short)0, acc[i][c],
                    false, false);
        __syncthreads();        // this buffer may be refilled next iteration
    }
#else
    // ---------------- register-prefetch pipeline (fallback) ----------------
    __shared__ __align__(16) _Float16 As[128][32];
    __shared__ __align__(16) _Float16 Bs[128][32];

    uint4 ra0 = *(const uint4*)(aptr);
    uint4 ra1 = *(const uint4*)(aptr + 8);
    uint4 rb0 = *(const uint4*)(bptr);
    uint4 rb1 = *(const uint4*)(bptr + 8);

    for (int k0 = 0; k0 < K; k0 += 32) {
        __syncthreads();
        *(uint4*)&As[srow][scol]     = ra0;
        *(uint4*)&As[srow][scol + 8] = ra1;
        *(uint4*)&Bs[srow][scol]     = rb0;
        *(uint4*)&Bs[srow][scol + 8] = rb1;
        __syncthreads();

        if (k0 + 32 < K) {
            ra0 = *(const uint4*)(aptr + k0 + 32);
            ra1 = *(const uint4*)(aptr + k0 + 40);
            rb0 = *(const uint4*)(bptr + k0 + 32);
            rb1 = *(const uint4*)(bptr + k0 + 40);
        }

        Frag af[2];
        #pragma unroll
        for (int i = 0; i < 2; ++i) {
            const int row = wm * 32 + i * 16 + ln;
            af[i].q[0] = *(const uint4*)&As[row][8 * hi];
            af[i].q[1] = *(const uint4*)&As[row][16 + 8 * hi];
        }
        Frag bf[4];
        #pragma unroll
        for (int c = 0; c < 4; ++c) {
            const int n = wn * 64 + c * 16 + ln;
            bf[c].q[0] = *(const uint4*)&Bs[n][8 * hi];
            bf[c].q[1] = *(const uint4*)&Bs[n][16 + 8 * hi];
        }
        #pragma unroll
        for (int i = 0; i < 2; ++i)
            #pragma unroll
            for (int c = 0; c < 4; ++c)
                acc[i][c] = __builtin_amdgcn_wmma_f32_16x16x32_f16(
                    false, af[i].v, false, bf[c].v, (short)0, acc[i][c],
                    false, false);
    }
#endif

    // Epilogue. D layout: element (r,lane): m = r + 8*hi, n = ln
    #pragma unroll
    for (int i = 0; i < 2; ++i) {
        #pragma unroll
        for (int c = 0; c < 4; ++c) {
            #pragma unroll
            for (int r = 0; r < 8; ++r) {
                const int gm = m0 + wm * 32 + i * 16 + r + 8 * hi;
                const int gn = n0 + wn * 64 + c * 16 + ln;
                const float v = acc[i][c][r] + bias[gn];
                if (outH) outH[(size_t)gm * N + gn] = (_Float16)v;
                else      outF[(size_t)gm * N + gn] = v;
            }
        }
    }
}

// ---------------------------------------------------------------------------
// Flash attention with temporal decay bias.
// grid: (SEQ/64, BATCH*N_HEADS), block: 128 threads = 4 waves.
// Each wave owns 16 query rows x all 64 keys of the tile; online softmax
// reductions via shfl within 16-lane halves. Next K/V tile prefetched into
// registers while current tile computes.
// ---------------------------------------------------------------------------
__global__ __launch_bounds__(128)
void ta_flash_attn(const _Float16* __restrict__ qkv,
                   const float* __restrict__ ts,
                   const float* __restrict__ maskp,
                   const float* __restrict__ decay_rate,
                   _Float16* __restrict__ attn_out) {
    __shared__ __align__(16) _Float16 Kt[64][64];   // [key][d]
    __shared__ __align__(16) _Float16 Vt[64][64];   // [d][key] (transposed)
    __shared__ __align__(16) _Float16 Pt[64][64];   // [qrow][key], per-wave rows
    __shared__ float tks[64];
    __shared__ float mks[64];

    const int tid  = threadIdx.x;
    const int wave = tid >> 5;
    const int lane = tid & 31;
    const int ln   = lane & 15;
    const int hi   = lane >> 4;

    const int bh = blockIdx.y;
    const int b  = bh / N_HEADS;
    const int h  = bh % N_HEADS;
    const int qt = blockIdx.x;           // query tile (64 rows)

    const float scale = 0.125f;          // 1/sqrt(64)
    const float decay = log1pf(__expf(decay_rate[h])) * (1.0f / 24.0f);

    // staging assignment: row = tid/2 (key), cols (tid&1)*32 .. +31
    const int skey = tid >> 1;
    const int sd   = (tid & 1) * 32;

    // Per-lane row metadata: rows r + 8*hi of this wave's 16-row strip.
    float tq[8];
    int   qg[8];
    #pragma unroll
    for (int r = 0; r < 8; ++r) {
        qg[r] = qt * 64 + wave * 16 + r + 8 * hi;
        tq[r] = ts[(size_t)b * SEQ + qg[r]];
    }

    // Q fragments (A-operand), loaded once.
    Frag qf[2];
    {
        const int row = qt * 64 + wave * 16 + ln;
        const _Float16* qp =
            qkv + (size_t)(b * SEQ + row) * QKV_N + h * D_HEAD;
        #pragma unroll
        for (int ks = 0; ks < 2; ++ks) {
            qf[ks].q[0] = *(const uint4*)(qp + ks * 32 + 8 * hi);
            qf[ks].q[1] = *(const uint4*)(qp + ks * 32 + 16 + 8 * hi);
        }
    }

    v8f   O[4];
    float Mrow[8], Lrow[8];
    #pragma unroll
    for (int c = 0; c < 4; ++c) O[c] = (v8f)0.0f;
    #pragma unroll
    for (int r = 0; r < 8; ++r) { Mrow[r] = -1e30f; Lrow[r] = 0.0f; }

    // prefetch tile 0 into registers
    HalfPack8 rk[4], rv[4];
    float tval = 0.0f, mval = 0.0f;
    {
        const size_t base = (size_t)(b * SEQ + skey) * QKV_N + h * D_HEAD + sd;
        #pragma unroll
        for (int j = 0; j < 4; ++j) {
            rk[j].q = *(const uint4*)(qkv + base + D_MODEL + j * 8);
            rv[j].q = *(const uint4*)(qkv + base + 2 * D_MODEL + j * 8);
        }
        if (tid < 64) {
            tval = ts[(size_t)b * SEQ + tid];
            mval = maskp[(size_t)b * SEQ + tid];
        }
    }

    for (int kt = 0; kt <= qt; ++kt) {
        // ---- commit prefetched tile to LDS ----
        #pragma unroll
        for (int j = 0; j < 4; ++j)
            *(uint4*)&Kt[skey][sd + j * 8] = rk[j].q;
        #pragma unroll
        for (int j = 0; j < 4; ++j)
            #pragma unroll
            for (int e = 0; e < 8; ++e)
                Vt[sd + j * 8 + e][skey] = rv[j].h[e];   // transpose scatter
        if (tid < 64) { tks[tid] = tval; mks[tid] = mval; }
        __syncthreads();

        // ---- prefetch next tile while this one computes ----
        if (kt < qt) {
            const size_t base =
                (size_t)(b * SEQ + (kt + 1) * 64 + skey) * QKV_N +
                h * D_HEAD + sd;
            #pragma unroll
            for (int j = 0; j < 4; ++j) {
                rk[j].q = *(const uint4*)(qkv + base + D_MODEL + j * 8);
                rv[j].q = *(const uint4*)(qkv + base + 2 * D_MODEL + j * 8);
            }
            if (tid < 64) {
                tval = ts[(size_t)b * SEQ + (kt + 1) * 64 + tid];
                mval = maskp[(size_t)b * SEQ + (kt + 1) * 64 + tid];
            }
        }

        // ---- S = Q K^T : 4 column blocks of 16 keys ----
        v8f sacc[4];
        #pragma unroll
        for (int c = 0; c < 4; ++c) {
            sacc[c] = (v8f)0.0f;
            #pragma unroll
            for (int ks = 0; ks < 2; ++ks) {
                Frag kf;  // B-operand: n = key, k = d (table order in d)
                const int key = c * 16 + ln;
                kf.q[0] = *(const uint4*)&Kt[key][ks * 32 + 8 * hi];
                kf.q[1] = *(const uint4*)&Kt[key][ks * 32 + 16 + 8 * hi];
                sacc[c] = __builtin_amdgcn_wmma_f32_16x16x32_f16(
                    false, qf[ks].v, false, kf.v, (short)0, sacc[c],
                    false, false);
            }
        }

        // ---- bias + masks + running row max ----
        float rmax[8];
        #pragma unroll
        for (int r = 0; r < 8; ++r) rmax[r] = -1e30f;
        #pragma unroll
        for (int c = 0; c < 4; ++c) {
            const int   kloc = c * 16 + ln;
            const int   kglb = kt * 64 + kloc;
            const float tk   = tks[kloc];
            const float mk   = mks[kloc];
            #pragma unroll
            for (int r = 0; r < 8; ++r) {
                float v = sacc[c][r] * scale - decay * fabsf(tq[r] - tk);
                if (kglb > qg[r] || mk == 0.0f) v = -1e30f;
                sacc[c][r] = v;
                rmax[r] = fmaxf(rmax[r], v);
            }
        }
        float alpha[8];
        #pragma unroll
        for (int r = 0; r < 8; ++r) {
            const float rm = rowmax16(rmax[r]);
            const float Mn = fmaxf(Mrow[r], rm);
            alpha[r] = __expf(Mrow[r] - Mn);
            Mrow[r]  = Mn;
            Lrow[r] *= alpha[r];
        }

        // ---- P = exp(S - M) -> per-wave LDS strip, row sums ----
        float rsum[8];
        #pragma unroll
        for (int r = 0; r < 8; ++r) rsum[r] = 0.0f;
        #pragma unroll
        for (int c = 0; c < 4; ++c) {
            #pragma unroll
            for (int r = 0; r < 8; ++r) {
                const float e = __expf(sacc[c][r] - Mrow[r]);
                rsum[r] += e;
                Pt[wave * 16 + r + 8 * hi][c * 16 + ln] = (_Float16)e;
            }
        }
        #pragma unroll
        for (int r = 0; r < 8; ++r) Lrow[r] += rowsum16(rsum[r]);

        // ---- rescale O, then O += P V ----
        #pragma unroll
        for (int c = 0; c < 4; ++c) {
            #pragma unroll
            for (int r = 0; r < 8; ++r) O[c][r] *= alpha[r];
        }
        #pragma unroll
        for (int ks = 0; ks < 2; ++ks) {
            Frag pf;  // A-operand from own wave's Pt strip
            const int row = wave * 16 + ln;
            pf.q[0] = *(const uint4*)&Pt[row][ks * 32 + 8 * hi];
            pf.q[1] = *(const uint4*)&Pt[row][ks * 32 + 16 + 8 * hi];
            #pragma unroll
            for (int c = 0; c < 4; ++c) {
                Frag vf;  // B-operand: n = d, k = key (table order via Vt)
                const int d = c * 16 + ln;
                vf.q[0] = *(const uint4*)&Vt[d][ks * 32 + 8 * hi];
                vf.q[1] = *(const uint4*)&Vt[d][ks * 32 + 16 + 8 * hi];
                O[c] = __builtin_amdgcn_wmma_f32_16x16x32_f16(
                    false, pf.v, false, vf.v, (short)0, O[c], false, false);
            }
        }
        __syncthreads();  // Kt/Vt free for next commit
    }

    // ---- normalize and store f16 to (B, L, H*dh) layout ----
    #pragma unroll
    for (int c = 0; c < 4; ++c) {
        #pragma unroll
        for (int r = 0; r < 8; ++r) {
            const float o = O[c][r] / Lrow[r];
            attn_out[(size_t)(b * SEQ + qg[r]) * D_MODEL +
                     h * D_HEAD + c * 16 + ln] = (_Float16)o;
        }
    }
}

// ---------------------------------------------------------------------------
// Host-side launcher
// ---------------------------------------------------------------------------
extern "C" void kernel_launch(void* const* d_in, const int* in_sizes, int n_in,
                              void* d_out, int out_size, void* d_ws, size_t ws_size,
                              hipStream_t stream) {
    const float* x     = (const float*)d_in[0];
    const float* ts    = (const float*)d_in[1];
    const float* maskp = (const float*)d_in[2];
    const float* W_qkv = (const float*)d_in[3];
    const float* b_qkv = (const float*)d_in[4];
    const float* W_out = (const float*)d_in[5];
    const float* b_out = (const float*)d_in[6];
    const float* decay = (const float*)d_in[7];
    float* out = (float*)d_out;

    // Workspace layout (f16 buffers)
    char* ws = (char*)d_ws;
    size_t off = 0;
    auto alloc = [&](size_t bytes) {
        char* p = ws + off;
        off += (bytes + 255) & ~(size_t)255;
        return p;
    };
    _Float16* xh   = (_Float16*)alloc((size_t)TOKENS * D_MODEL * 2);  // 16 MB
    _Float16* wqh  = (_Float16*)alloc((size_t)QKV_N * D_MODEL * 2);   //  6 MB
    _Float16* woh  = (_Float16*)alloc((size_t)D_MODEL * D_MODEL * 2); //  2 MB
    _Float16* qkvh = (_Float16*)alloc((size_t)TOKENS * QKV_N * 2);    // 48 MB
    _Float16* aoh  = (_Float16*)alloc((size_t)TOKENS * D_MODEL * 2);  // 16 MB
    (void)ws_size;

    // 1) casts (n/4 threads, vectorized)
    {
        int n = TOKENS * D_MODEL / 4;
        ta_cast_f32_f16<<<(n + 255) / 256, 256, 0, stream>>>(x, xh, n);
        n = QKV_N * D_MODEL / 4;
        ta_cast_f32_f16<<<(n + 255) / 256, 256, 0, stream>>>(W_qkv, wqh, n);
        n = D_MODEL * D_MODEL / 4;
        ta_cast_f32_f16<<<(n + 255) / 256, 256, 0, stream>>>(W_out, woh, n);
    }

    // 2) qkv = x @ W_qkv^T + b_qkv   (f16 out)
    {
        dim3 grid(QKV_N / 128, TOKENS / 128);
        ta_gemm_f16<<<grid, 256, 0, stream>>>(
            xh, wqh, b_qkv, nullptr, qkvh, TOKENS, QKV_N, D_MODEL);
    }

    // 3) flash attention
    {
        dim3 grid(SEQ / 64, BATCH * N_HEADS);
        ta_flash_attn<<<grid, 128, 0, stream>>>(qkvh, ts, maskp, decay, aoh);
    }

    // 4) out = attn_out @ W_out^T + b_out  (f32 out)
    {
        dim3 grid(D_MODEL / 128, TOKENS / 128);
        ta_gemm_f16<<<grid, 256, 0, stream>>>(
            aoh, woh, b_out, out, nullptr, TOKENS, D_MODEL, D_MODEL);
    }
}
